// Cross_Former_87720412053645
// MI455X (gfx1250) — compile-verified, matrix operand order checked by reading, not run on hardware
//
#include <hip/hip_runtime.h>
#include <hip/hip_bf16.h>
#include <math.h>

// ---------------- problem constants ----------------
#define NND   2048      // N nodes
#define FIN   500
#define FPAD  512
#define HD    128       // H
#define LL    2
#define CCLS  10
#define CPAD  32        // classifier N padded to one 32-wide tile pair
#define FFND  256
#define EE    32768
static const size_t NN = (size_t)NND * (size_t)NND;

// ---------------- WMMA vector types ----------------
typedef __attribute__((ext_vector_type(16))) __bf16          v16bf;
typedef __attribute__((ext_vector_type(16))) unsigned short  v16u;
typedef __attribute__((ext_vector_type(8)))  unsigned short  v8u;
typedef __attribute__((ext_vector_type(8)))  float           v8f;

__device__ __forceinline__ unsigned short f2bf(float f) {
  unsigned int u = __float_as_uint(f);
  u += 0x7FFFu + ((u >> 16) & 1u);          // round to nearest even
  return (unsigned short)(u >> 16);
}

// ---------------- aligned bf16 WMMA GEMM:  C[M,N] = A[M,K] * Bt[N,K]^T ----------------
// Requirements: M%32==0, N%32==0, K%32==0; A,Bt bf16 row-major, 16B-aligned rows.
// flags: 2=store bf16, 4=exact GELU, 8=+bias[n], 16=store transposed (C[n*ldc+m])
#define GF_OUTBF  2
#define GF_GELU   4
#define GF_BIAS   8
#define GF_STORET 16

__device__ __forceinline__ v16bf ld_k32(const unsigned short* p) {
  // lane's 16 bf16: two contiguous 8-element runs at +0 and +16 elements
  v8u lo = *(const v8u*)p;
  v8u hi = *(const v8u*)(p + 16);
  return __builtin_bit_cast(v16bf, __builtin_shufflevector(
      lo, hi, 0,1,2,3,4,5,6,7,8,9,10,11,12,13,14,15));
}

__global__ void wmma_gemm_nt(const unsigned short* __restrict__ A,
                             const unsigned short* __restrict__ Bt,
                             const float* __restrict__ bias,
                             void* __restrict__ Cout,
                             int M, int N, int K,
                             int lda, int ldb, int ldc, int flags)
{
  const int wave   = (int)((blockIdx.x * blockDim.x + threadIdx.x) >> 5);
  const int lane   = (int)(threadIdx.x & 31);
  const int tilesN = N >> 5;
  if (wave >= (M >> 5) * tilesN) return;    // uniform: whole wave exits together
  const int m0 = (wave / tilesN) << 5;
  const int n0 = (wave % tilesN) << 5;

  // CDNA5 16-bit A/B operand layout: lane L(0..15) holds row L, K in [0,8)+[16,24);
  // lanes 16..31 hold same rows, K in [8,16)+[24,32).
  const int r16   = lane & 15;
  const int khalf = (lane >> 4) << 3;

  const unsigned short* pA0 = A  + (size_t)(m0 + r16) * lda + khalf;
  const unsigned short* pA1 = pA0 + (size_t)16 * lda;
  const unsigned short* pB0 = Bt + (size_t)(n0 + r16) * ldb + khalf;
  const unsigned short* pB1 = pB0 + (size_t)16 * ldb;

  v8f z = {0.f,0.f,0.f,0.f,0.f,0.f,0.f,0.f};
  v8f acc00 = z, acc01 = z, acc10 = z, acc11 = z;

  for (int k0 = 0; k0 < K; k0 += 32) {
    v16bf a0 = ld_k32(pA0), a1 = ld_k32(pA1);
    v16bf b0 = ld_k32(pB0), b1 = ld_k32(pB1);
    pA0 += 32; pA1 += 32; pB0 += 32; pB1 += 32;
    acc00 = __builtin_amdgcn_wmma_f32_16x16x32_bf16(false, a0, false, b0, (short)0, acc00, false, false);
    acc01 = __builtin_amdgcn_wmma_f32_16x16x32_bf16(false, a0, false, b1, (short)0, acc01, false, false);
    acc10 = __builtin_amdgcn_wmma_f32_16x16x32_bf16(false, a1, false, b0, (short)0, acc10, false, false);
    acc11 = __builtin_amdgcn_wmma_f32_16x16x32_bf16(false, a1, false, b1, (short)0, acc11, false, false);
  }

  // C/D 16x16 f32 layout: VGPR r -> M = r + 8*(lane>>4), N = lane&15
  const int mrow = (lane >> 4) << 3;
  float bv0 = 0.f, bv1 = 0.f;
  if (flags & GF_BIAS) { bv0 = bias[n0 + r16]; bv1 = bias[n0 + 16 + r16]; }

  auto store8 = [&](v8f acc, int mb, int n, float bv) {
#pragma unroll
    for (int r = 0; r < 8; ++r) {
      float v = acc[r] + bv;
      if (flags & GF_GELU) v = 0.5f * v * (1.0f + erff(v * 0.70710678118654752f));
      const size_t idx = (flags & GF_STORET) ? ((size_t)n * ldc + (mb + r))
                                             : ((size_t)(mb + r) * ldc + n);
      if (flags & GF_OUTBF) ((unsigned short*)Cout)[idx] = f2bf(v);
      else                  ((float*)Cout)[idx] = v;
    }
  };
  store8(acc00, m0 + mrow,      n0 + r16,      bv0);
  store8(acc01, m0 + mrow,      n0 + 16 + r16, bv1);
  store8(acc10, m0 + 16 + mrow, n0 + r16,      bv0);
  store8(acc11, m0 + 16 + mrow, n0 + 16 + r16, bv1);
}

// ---------------- elementwise / small kernels ----------------
__global__ void k_zero_f32(float* p, int n) {
  int i = blockIdx.x * blockDim.x + threadIdx.x;
  if (i < n) p[i] = 0.f;
}
__global__ void k_build_adj(const int* __restrict__ ei, float* __restrict__ adj) {
  int e = blockIdx.x * blockDim.x + threadIdx.x;
  if (e < EE) atomicAdd(&adj[(size_t)ei[e] * NND + ei[EE + e]], 1.0f);
}
__global__ void k_f2bf(const float* __restrict__ s, unsigned short* __restrict__ d, int n) {
  int i = blockIdx.x * blockDim.x + threadIdx.x;
  if (i < n) d[i] = f2bf(s[i]);
}
__global__ void k_pad_rows(const float* __restrict__ s, unsigned short* __restrict__ d,
                           int rows, int scols, int dcols) {
  int i = blockIdx.x * blockDim.x + threadIdx.x;
  if (i < rows * dcols) {
    int r = i / dcols, c = i % dcols;
    d[i] = (c < scols) ? f2bf(s[(size_t)r * scols + c]) : (unsigned short)0;
  }
}
// weight transpose+convert: src f32 [R,C] -> dst bf16 [Cpad,Rpad], zero-padded
__global__ void k_wT(const float* __restrict__ s, unsigned short* __restrict__ d,
                     int R, int C, int Rpad, int Cpad) {
  int i = blockIdx.x * blockDim.x + threadIdx.x;
  if (i < Cpad * Rpad) {
    int n = i / Rpad, k = i % Rpad;
    d[i] = (n < C && k < R) ? f2bf(s[(size_t)k * C + n]) : (unsigned short)0;
  }
}
__global__ void k_pad_bias(const float* __restrict__ s, float* __restrict__ d, int n, int npad) {
  int i = blockIdx.x * blockDim.x + threadIdx.x;
  if (i < npad) d[i] = (i < n) ? s[i] : 0.f;
}
__global__ void k_copy_f32(const float* __restrict__ s, float* __restrict__ d, int n) {
  int i = blockIdx.x * blockDim.x + threadIdx.x;
  if (i < n) d[i] = s[i];
}
__global__ void k_add_f32(float* __restrict__ d, const float* __restrict__ s, int n) {
  int i = blockIdx.x * blockDim.x + threadIdx.x;
  if (i < n) d[i] += s[i];
}
__global__ void k_f2bf_strided(const float* __restrict__ s, unsigned short* __restrict__ d,
                               int rows, int cols, int ldd, int coff) {
  int i = blockIdx.x * blockDim.x + threadIdx.x;
  if (i < rows * cols) {
    int r = i / cols, c = i % cols;
    d[(size_t)r * ldd + coff + c] = f2bf(s[i]);
  }
}

// row softmax over W fp32 logits -> bf16 probabilities (one 256-thread block per row)
__global__ void k_softmax_row_bf16(const float* __restrict__ in,
                                   unsigned short* __restrict__ out, int W) {
  __shared__ float red[256];
  const int row = blockIdx.x, t = threadIdx.x;
  const float* rp = in + (size_t)row * W;
  float lmax = -3.4e38f;
  for (int c = t; c < W; c += 256) lmax = fmaxf(lmax, rp[c]);
  red[t] = lmax; __syncthreads();
  for (int s = 128; s > 0; s >>= 1) { if (t < s) red[t] = fmaxf(red[t], red[t + s]); __syncthreads(); }
  const float m = red[0]; __syncthreads();
  float lsum = 0.f;
  for (int c = t; c < W; c += 256) lsum += __expf(rp[c] - m);
  red[t] = lsum; __syncthreads();
  for (int s = 128; s > 0; s >>= 1) { if (t < s) red[t] += red[t + s]; __syncthreads(); }
  const float inv = 1.0f / red[0];
  unsigned short* op = out + (size_t)row * W;
  for (int c = t; c < W; c += 256) op[c] = f2bf(__expf(rp[c] - m) * inv);
}

// LayerNorm over H=128 -> bf16 (optional transposed copy, relu, column offset)
__global__ void k_layernorm_bf16(const float* __restrict__ x, const float* __restrict__ scale,
                                 const float* __restrict__ bias, unsigned short* __restrict__ out,
                                 unsigned short* __restrict__ outT, int ldo, int coff, int relu) {
  __shared__ float red[HD];
  __shared__ float stats[2];
  const int row = blockIdx.x, t = threadIdx.x;   // 128 threads
  const float v = x[(size_t)row * HD + t];
  red[t] = v; __syncthreads();
  for (int s = 64; s > 0; s >>= 1) { if (t < s) red[t] += red[t + s]; __syncthreads(); }
  if (t == 0) stats[0] = red[0] * (1.0f / HD);
  __syncthreads();
  const float d = v - stats[0];
  red[t] = d * d; __syncthreads();
  for (int s = 64; s > 0; s >>= 1) { if (t < s) red[t] += red[t + s]; __syncthreads(); }
  if (t == 0) stats[1] = rsqrtf(red[0] * (1.0f / HD) + 1e-5f);
  __syncthreads();
  float y = d * stats[1] * scale[t] + bias[t];
  if (relu) y = fmaxf(y, 0.f);
  const unsigned short h = f2bf(y);
  out[(size_t)row * ldo + coff + t] = h;
  if (outT) outT[(size_t)t * NND + row] = h;
}

// gated two-branch mix: g=sigmoid([b1@av0, b2@av1]); att=softmax(g@av); x1 += att0*b1+att1*b2
__global__ void k_gate(float* __restrict__ x1, const float* __restrict__ b1,
                       const float* __restrict__ b2, const float* __restrict__ av0,
                       const float* __restrict__ av1, const float* __restrict__ av) {
  __shared__ float r0[HD], r1[HD];
  const int row = blockIdx.x, t = threadIdx.x;   // 128 threads
  const float vb1 = b1[(size_t)row * HD + t], vb2 = b2[(size_t)row * HD + t];
  r0[t] = vb1 * av0[t]; r1[t] = vb2 * av1[t];
  __syncthreads();
  for (int s = 64; s > 0; s >>= 1) { if (t < s) { r0[t] += r0[t + s]; r1[t] += r1[t + s]; } __syncthreads(); }
  const float g0 = 1.0f / (1.0f + __expf(-r0[0]));
  const float g1 = 1.0f / (1.0f + __expf(-r1[0]));
  const float a0 = g0 * av[0] + g1 * av[2];
  const float a1 = g0 * av[1] + g1 * av[3];
  const float mx = fmaxf(a0, a1);
  const float e0 = __expf(a0 - mx), e1 = __expf(a1 - mx);
  const float att0 = e0 / (e0 + e1), att1 = e1 / (e0 + e1);
  x1[(size_t)row * HD + t] += att0 * vb1 + att1 * vb2;
}

__global__ void k_log_softmax(const float* __restrict__ in, float* __restrict__ out) {
  int row = blockIdx.x * blockDim.x + threadIdx.x;
  if (row < NND) {
    const float* rp = in + (size_t)row * CPAD;   // padded logits, first CCLS valid
    float m = rp[0];
    for (int c = 1; c < CCLS; ++c) m = fmaxf(m, rp[c]);
    float s = 0.f;
    for (int c = 0; c < CCLS; ++c) s += __expf(rp[c] - m);
    const float ls = __logf(s);
    float* op = out + (size_t)row * CCLS;
    for (int c = 0; c < CCLS; ++c) op[c] = rp[c] - m - ls;
  }
}

// ---------------- launcher ----------------
extern "C" void kernel_launch(void* const* d_in, const int* in_sizes, int n_in,
                              void* d_out, int out_size, void* d_ws, size_t ws_size,
                              hipStream_t stream) {
  (void)in_sizes; (void)n_in; (void)out_size; (void)ws_size;
  const float* x        = (const float*)d_in[0];
  const int*   ei       = (const int*)d_in[1];
  const float* lin1_w   = (const float*)d_in[2];
  const float* lin1_b   = (const float*)d_in[3];
  const float* aln_s    = (const float*)d_in[4];
  const float* aln_b    = (const float*)d_in[5];
  const float* layer_w  = (const float*)d_in[6];
  const float* layer_b  = (const float*)d_in[7];
  const float* out_w    = (const float*)d_in[8];
  const float* out_b    = (const float*)d_in[9];
  const float* av0      = (const float*)d_in[10];
  const float* av1      = (const float*)d_in[11];
  const float* av       = (const float*)d_in[12];
  const float* fln_s    = (const float*)d_in[13];
  const float* fln_b    = (const float*)d_in[14];
  const float* ffn1_w   = (const float*)d_in[15];
  const float* ffn1_b   = (const float*)d_in[16];
  const float* ffn2_w   = (const float*)d_in[17];
  const float* ffn2_b   = (const float*)d_in[18];
  const float* finln_s  = (const float*)d_in[19];
  const float* finln_b  = (const float*)d_in[20];
  const float* cls_w    = (const float*)d_in[21];
  const float* cls_b    = (const float*)d_in[22];
  float* outp = (float*)d_out;

  // ---- workspace layout ----
  char* w = (char*)d_ws;
  size_t off = 0;
  auto take = [&](size_t bytes) -> char* {
    char* p = w + off; off = (off + bytes + 255) & ~(size_t)255; return p;
  };
  unsigned short* adj_bf  = (unsigned short*)take(NN * 2);
  float*          nnf     = (float*)take(NN * 4);      // adj f32 -> AAT logits -> s logits
  unsigned short* AAT_bf  = (unsigned short*)take(NN * 2);
  unsigned short* s_bf    = (unsigned short*)take(NN * 2);
  unsigned short* xpad_bf = (unsigned short*)take((size_t)NND * FPAD * 2);
  unsigned short* lin1_t  = (unsigned short*)take((size_t)HD * FPAD * 2);       // [128,512]
  unsigned short* lw_t    = (unsigned short*)take((size_t)LL * HD * NND * 2);   // [128,2048] x L
  unsigned short* ow_t    = (unsigned short*)take((size_t)LL * HD * HD * 2);
  unsigned short* f1_t    = (unsigned short*)take((size_t)LL * FFND * HD * 2);  // [256,128] x L
  unsigned short* f2_t    = (unsigned short*)take((size_t)LL * HD * FFND * 2);  // [128,256] x L
  unsigned short* cls_t   = (unsigned short*)take((size_t)CPAD * (2*HD) * 2);   // [32,256]
  float*          clsb_p  = (float*)take(CPAD * 4);
  float*          x0      = (float*)take((size_t)NND * HD * 4);
  float*          x1      = (float*)take((size_t)NND * HD * 4);
  unsigned short* y_bf    = (unsigned short*)take((size_t)NND * HD * 2);
  unsigned short* yT_bf   = (unsigned short*)take((size_t)HD * NND * 2);
  unsigned short* tT_bf   = (unsigned short*)take((size_t)HD * NND * 2);  // (s@layer_w)^T
  unsigned short* t2_bf   = (unsigned short*)take((size_t)NND * HD * 2);  // AAT@y
  unsigned short* z_bf    = (unsigned short*)take((size_t)NND * HD * 2);
  float*          b1      = (float*)take((size_t)NND * HD * 4);
  float*          b2      = (float*)take((size_t)NND * HD * 4);
  unsigned short* fh_bf   = (unsigned short*)take((size_t)NND * FFND * 2);
  float*          z2      = (float*)take((size_t)NND * HD * 4);
  unsigned short* h_bf    = (unsigned short*)take((size_t)NND * 2 * HD * 2);
  float*          logits  = (float*)take((size_t)NND * CPAD * 4);

  auto gemm = [&](const unsigned short* A, const unsigned short* Bt, const float* bias,
                  void* C, int M, int N, int K, int lda, int ldb, int ldc, int flags) {
    int tiles = (M >> 5) * (N >> 5);         // 32x32 tile per wave
    int blocks = (tiles + 7) / 8;            // 8 waves (256 thr) per block
    wmma_gemm_nt<<<blocks, 256, 0, stream>>>(A, Bt, bias, C, M, N, K, lda, ldb, ldc, flags);
  };
  auto gb = [](int n) { return (n + 255) / 256; };

  // ---- weight prep: convert + transpose to [N,K] bf16 (padded where needed) ----
  k_pad_rows<<<gb(NND*FPAD), 256, 0, stream>>>(x, xpad_bf, NND, FIN, FPAD);
  k_wT<<<gb(HD*FPAD), 256, 0, stream>>>(lin1_w, lin1_t, FIN, HD, FPAD, HD);
  for (int l = 0; l < LL; ++l) {
    k_wT<<<gb(HD*NND), 256, 0, stream>>>(layer_w + (size_t)l*NND*HD, lw_t + (size_t)l*HD*NND, NND, HD, NND, HD);
    k_wT<<<gb(HD*HD), 256, 0, stream>>>(out_w + (size_t)l*HD*HD, ow_t + (size_t)l*HD*HD, HD, HD, HD, HD);
    k_wT<<<gb(FFND*HD), 256, 0, stream>>>(ffn1_w + (size_t)l*HD*FFND, f1_t + (size_t)l*FFND*HD, HD, FFND, HD, FFND);
    k_wT<<<gb(HD*FFND), 256, 0, stream>>>(ffn2_w + (size_t)l*FFND*HD, f2_t + (size_t)l*HD*FFND, FFND, HD, FFND, HD);
  }
  k_wT<<<gb(CPAD*2*HD), 256, 0, stream>>>(cls_w, cls_t, 2*HD, CCLS, 2*HD, CPAD);
  k_pad_bias<<<1, CPAD, 0, stream>>>(cls_b, clsb_p, CCLS, CPAD);

  // ---- adjacency (integer-valued -> exact in bf16) ----
  k_zero_f32<<<gb((int)NN), 256, 0, stream>>>(nnf, (int)NN);
  k_build_adj<<<gb(EE), 256, 0, stream>>>(ei, nnf);
  k_f2bf<<<gb((int)NN), 256, 0, stream>>>(nnf, adj_bf, (int)NN);

  // ---- AAT = softmax(adj @ adj^T)  (Bt = adj itself) ----
  gemm(adj_bf, adj_bf, nullptr, nnf, NND, NND, NND, NND, NND, NND, 0);
  k_softmax_row_bf16<<<NND, 256, 0, stream>>>(nnf, AAT_bf, NND);

  // ---- x0 = x @ lin1_w + b ; x1 = x0 ----
  gemm(xpad_bf, lin1_t, lin1_b, x0, NND, HD, FPAD, FPAD, FPAD, HD, GF_BIAS);
  k_copy_f32<<<gb(NND*HD), 256, 0, stream>>>(x0, x1, NND*HD);

  // ---- layers ----
  for (int i = 0; i < LL; ++i) {
    // y = LN(x1): emit y [2048,128] and y^T [128,2048]
    k_layernorm_bf16<<<NND, HD, 0, stream>>>(x1, aln_s + i*HD, aln_b + i*HD, y_bf, yT_bf, HD, 0, 0);
    // s = softmax(y @ y^T)  (Bt = y itself)
    gemm(y_bf, y_bf, nullptr, nnf, NND, NND, HD, HD, HD, NND, 0);
    k_softmax_row_bf16<<<NND, 256, 0, stream>>>(nnf, s_bf, NND);
    // b1 = adj @ (s @ layer_w) + layer_b   (associativity: ~8.5x fewer FLOPs)
    gemm(s_bf, lw_t + (size_t)i*HD*NND, nullptr, tT_bf,
         NND, HD, NND, NND, NND, NND, GF_OUTBF | GF_STORET);    // store t^T [128,2048]
    gemm(adj_bf, tT_bf, layer_b + i*HD, b1, NND, HD, NND, NND, NND, HD, GF_BIAS);
    // b2 = (AAT @ y) @ out_w + out_b
    gemm(AAT_bf, yT_bf, nullptr, t2_bf, NND, HD, NND, NND, NND, HD, GF_OUTBF);
    gemm(t2_bf, ow_t + (size_t)i*HD*HD, out_b + i*HD, b2, NND, HD, HD, HD, HD, HD, GF_BIAS);
    // gated residual mix
    k_gate<<<NND, HD, 0, stream>>>(x1, b1, b2, av0 + i*HD, av1 + i*HD, av + i*4);
    // FFN
    k_layernorm_bf16<<<NND, HD, 0, stream>>>(x1, fln_s + i*HD, fln_b + i*HD, z_bf, nullptr, HD, 0, 0);
    gemm(z_bf, f1_t + (size_t)i*FFND*HD, ffn1_b + i*FFND, fh_bf,
         NND, FFND, HD, HD, HD, FFND, GF_BIAS | GF_GELU | GF_OUTBF);
    gemm(fh_bf, f2_t + (size_t)i*HD*FFND, ffn2_b + i*HD, z2,
         NND, HD, FFND, FFND, FFND, HD, GF_BIAS);
    k_add_f32<<<gb(NND*HD), 256, 0, stream>>>(x1, z2, NND*HD);
  }

  // ---- final: h = [x0, relu(LN(x1))]; out = log_softmax(h @ cls_w + b) ----
  k_layernorm_bf16<<<NND, HD, 0, stream>>>(x1, finln_s, finln_b, h_bf, nullptr, 2*HD, HD, 1);
  k_f2bf_strided<<<gb(NND*HD), 256, 0, stream>>>(x0, h_bf, NND, HD, 2*HD, 0);
  gemm(h_bf, cls_t, clsb_p, logits, NND, CPAD, 2*HD, 2*HD, 2*HD, CPAD, GF_BIAS);
  k_log_softmax<<<gb(NND), 256, 0, stream>>>(logits, outp);
}
